// MultiHeadLatentAttention_44092134260971
// MI455X (gfx1250) — matrix operand order repaired and checked by reading, not compiled, not run on hardware
//
#include <hip/hip_runtime.h>
#include <hip/hip_bf16.h>

// ---------------------------------------------------------------------------
// MLA forward for MI455X (gfx1250, wave32, WMMA bf16 16x16x32, f32 accum).
//   1) convert x / weights fp32 -> bf16
//   2) q   = x @ w_query          (WMMA GEMM)
//   3) kv  = x @ wkv_a            (WMMA GEMM)
//   4) kvn = rmsnorm(kv[:,:512])  -> bf16
//   5) dec = kvn @ wkv_b          (WMMA GEMM)
//   6) pack: Q (rope+scale), K (nope|rope), V^T  -> bf16 per-head tensors
//   7) flash-style causal attention (WMMA QK^T and PV, online softmax)
//   8) out = attn_out @ out_proj_w (WMMA GEMM, f32 -> d_out)
// A-tile / Q / K / V^T staging uses the Tensor Data Mover when the builtin is
// available (TENSORcnt + LDS pad feature); falls back to manual staging.
// ---------------------------------------------------------------------------

typedef __attribute__((ext_vector_type(16))) __bf16 v16bf;
typedef __attribute__((ext_vector_type(8)))  float  v8f;

#if defined(__gfx1250__) && __has_builtin(__builtin_amdgcn_tensor_load_to_lds) && \
    __has_builtin(__builtin_amdgcn_s_wait_tensorcnt)
#define USE_TDM 1
#else
#define USE_TDM 0
#endif

__device__ __forceinline__ unsigned short f32_bf16(float f) {
    union { float f; unsigned int u; } x; x.f = f;
    unsigned int u = x.u;
    unsigned int r = u + 0x7FFFu + ((u >> 16) & 1u);   // round-to-nearest-even
    if ((u & 0x7F800000u) == 0x7F800000u) r = u;       // inf/nan passthrough
    return (unsigned short)(r >> 16);
}

__device__ __forceinline__ void put2(v16bf& f, int j, unsigned int u) {
    union { unsigned int u; __bf16 h[2]; } cv; cv.u = u;
    f[2 * j]     = cv.h[0];
    f[2 * j + 1] = cv.h[1];
}

#if USE_TDM
typedef __attribute__((ext_vector_type(4))) unsigned int u32x4;
typedef __attribute__((ext_vector_type(8))) int          i32x8;
typedef __attribute__((ext_vector_type(4))) int          i32x4;

__device__ __forceinline__ unsigned lds_byte_off(const void* p) {
    // generic -> LDS(AS3) -> 32-bit LDS byte offset
    return (unsigned)(unsigned long long)(__attribute__((address_space(3))) const void*)p;
}

// 2D tile DMA: global (row stride stride0 elems, 2-byte elems) -> LDS, with
// per-row LDS padding (pad_interval code: 2<<c dwords per row; pad 1 dword).
__device__ __forceinline__ void tdm_load_2d(const void* gaddr, unsigned lds_addr,
                                            unsigned tile_d0, unsigned tile_d1,
                                            unsigned tensor_d0, unsigned tensor_d1,
                                            unsigned long long stride0,
                                            unsigned pad_code) {
    unsigned long long ga = (unsigned long long)gaddr;
    u32x4 g0;
    g0[0] = 1u;                                            // count=1, user D#
    g0[1] = lds_addr;                                      // LDS byte address
    g0[2] = (unsigned)(ga & 0xFFFFFFFFu);                  // global addr lo
    g0[3] = (unsigned)((ga >> 32) & 0x1FFFFFFu) | (2u << 30);  // hi | type=2
    i32x8 g1;
    g1[0] = (int)((1u << 16)                               // data_size = 2B
                 | (1u << 20)                              // pad_enable
                 | (pad_code << 22)                        // pad_interval
                 | (0u << 25));                            // pad_amount = 1 dw
    g1[1] = (int)((tensor_d0 & 0xFFFFu) << 16);            // dim0[15:0]
    g1[2] = (int)((tensor_d0 >> 16) | ((tensor_d1 & 0xFFFFu) << 16));
    g1[3] = (int)((tensor_d1 >> 16) | (tile_d0 << 16));
    g1[4] = (int)(tile_d1 & 0xFFFFu);                      // tile_dim1 (dim2=0)
    g1[5] = (int)(unsigned)(stride0 & 0xFFFFFFFFu);        // dim0 stride lo
    g1[6] = (int)(unsigned)((stride0 >> 32) & 0xFFFFu);    // stride hi | s1 lo
    g1[7] = 0;
    i32x4 z4 = {0, 0, 0, 0};
    i32x8 z8 = {0, 0, 0, 0, 0, 0, 0, 0};
    __builtin_amdgcn_tensor_load_to_lds(g0, g1, z4, z4, z8, 0);
}
#endif

// ---------------------------------------------------------------------------
// fp32 -> bf16 elementwise convert
// ---------------------------------------------------------------------------
__global__ void cvt_bf16(const float* __restrict__ in, unsigned short* __restrict__ out,
                         size_t n) {
    size_t i = (size_t)blockIdx.x * blockDim.x + threadIdx.x;
    if (i < n) out[i] = f32_bf16(in[i]);
}

// ---------------------------------------------------------------------------
// Generic bf16 WMMA GEMM: C(MxN,f32) = A(MxK,bf16 rowmaj) @ B(KxN,bf16 rowmaj)
// Block 256 thr (8 waves), tile 128x128x32; wave grid 4(M)x2(N) of 2x4 WMMAs.
// A tile staged by TDM (wave 0) when available; B staged transposed manually.
// ---------------------------------------------------------------------------
#define G_BM 128
#define G_BN 128
#define G_BK 32
#define G_LDA 17   // dword stride of padded 32-wide bf16 row
#define G_LDB 17

__launch_bounds__(256)
__global__ void gemm_bf16_wmma(const unsigned short* __restrict__ A,
                               const unsigned short* __restrict__ B,
                               float* __restrict__ C,
                               int M, int N, int K) {
    __shared__ unsigned int sA[G_BM * G_LDA];   // [m][k] row-major, padded
    __shared__ unsigned int sB[G_BN * G_LDB];   // [n][k] (transposed), padded

    const int t  = threadIdx.x;
    const int m0 = blockIdx.y * G_BM;
    const int n0 = blockIdx.x * G_BN;

    const int wv  = t >> 5;
    const int ln  = t & 31;
    const int wm  = wv & 3;
    const int wn  = wv >> 2;
    const int lg  = ln >> 4;
    const int l16 = ln & 15;

    v8f acc[2][4];
#pragma unroll
    for (int i = 0; i < 2; ++i)
#pragma unroll
        for (int j = 0; j < 4; ++j)
#pragma unroll
            for (int r = 0; r < 8; ++r) acc[i][j][r] = 0.0f;

    for (int kb = 0; kb < K; kb += G_BK) {
        // ---- stage A tile: 128x32 bf16 ----
#if USE_TDM
        if (wv == 0) {
            tdm_load_2d(A + (size_t)m0 * K + kb, lds_byte_off(sA),
                        /*tile*/ G_BK, G_BM,
                        /*tensor*/ (unsigned)(K - kb), (unsigned)(M - m0),
                        (unsigned long long)K, /*pad: 16 dw rows*/ 3u);
        }
#else
#pragma unroll
        for (int it = 0; it < 2; ++it) {
            int ch = it * 256 + t;
            int r  = ch >> 2;
            int c8 = (ch & 3) * 8;
            uint4 d = make_uint4(0u, 0u, 0u, 0u);
            if (m0 + r < M)
                d = *(const uint4*)(A + (size_t)(m0 + r) * K + kb + c8);
            unsigned int* s = &sA[r * G_LDA + (c8 >> 1)];
            s[0] = d.x; s[1] = d.y; s[2] = d.z; s[3] = d.w;
        }
#endif
        // ---- stage B tile: 32x128 bf16, store transposed [n][k] ----
#pragma unroll
        for (int it = 0; it < 2; ++it) {
            int ch = it * 256 + t;
            int kr = ch >> 4;
            int c8 = (ch & 15) * 8;
            int n  = n0 + c8;
            unsigned int v4[4] = {0u, 0u, 0u, 0u};
            const unsigned short* bp = B + (size_t)(kb + kr) * N + n;
            if (n + 7 < N) {
                uint4 dd = *(const uint4*)bp;
                v4[0] = dd.x; v4[1] = dd.y; v4[2] = dd.z; v4[3] = dd.w;
            } else {
#pragma unroll
                for (int j = 0; j < 8; ++j)
                    if (n + j < N) v4[j >> 1] |= ((unsigned int)bp[j]) << ((j & 1) * 16);
            }
            unsigned short* sb16 = (unsigned short*)sB;
#pragma unroll
            for (int j = 0; j < 8; ++j)
                sb16[(c8 + j) * (2 * G_LDB) + kr] =
                    (unsigned short)(v4[j >> 1] >> ((j & 1) * 16));
        }
        if (kb + G_BK < K) {
            __builtin_prefetch(A + (size_t)(m0 + (t & 127)) * K + kb + G_BK, 0, 1);
            __builtin_prefetch(B + (size_t)(kb + G_BK + (t & 31)) * N + n0, 0, 1);
        }
#if USE_TDM
        if (wv == 0) __builtin_amdgcn_s_wait_tensorcnt(0);
#endif
        __syncthreads();

        // ---- fragments + 8 WMMAs per wave ----
        v16bf af[2];
#pragma unroll
        for (int tm = 0; tm < 2; ++tm) {
            int m = wm * 32 + tm * 16 + l16;
            const unsigned int* base = &sA[m * G_LDA + lg * 8];
#pragma unroll
            for (int j = 0; j < 8; ++j) {
                int kd = j;
                put2(af[tm], j, base[kd]);
            }
        }
        v16bf bfr[4];
#pragma unroll
        for (int tn = 0; tn < 4; ++tn) {
            int n = wn * 64 + tn * 16 + l16;
            const unsigned int* base = &sB[n * G_LDB + lg * 8];
#pragma unroll
            for (int j = 0; j < 8; ++j) put2(bfr[tn], j, base[j]);
        }
#pragma unroll
        for (int tm = 0; tm < 2; ++tm)
#pragma unroll
            for (int tn = 0; tn < 4; ++tn)
                acc[tm][tn] = __builtin_amdgcn_wmma_f32_16x16x32_bf16(
                    false, af[tm], false, bfr[tn], (short)0, acc[tm][tn], false, false);
        __syncthreads();
    }

    // ---- store C (f32) ----
#pragma unroll
    for (int tm = 0; tm < 2; ++tm)
#pragma unroll
        for (int tn = 0; tn < 4; ++tn) {
            int row0 = m0 + wm * 32 + tm * 16 + lg * 8;
            int col  = n0 + wn * 64 + tn * 16 + l16;
            if (col < N) {
#pragma unroll
                for (int r = 0; r < 8; ++r) {
                    int row = row0 + r;
                    if (row < M) C[(size_t)row * N + col] = acc[tm][tn][r];
                }
            }
        }
}

// ---------------------------------------------------------------------------
// RMS-norm of kv[:, :512] -> bf16 (times kv_norm_w). One row per block.
// ---------------------------------------------------------------------------
__launch_bounds__(256)
__global__ void rmsnorm_kv(const float* __restrict__ kv, const float* __restrict__ w,
                           unsigned short* __restrict__ out) {
    const int row = blockIdx.x;
    const float* x = kv + (size_t)row * 576;
    __shared__ float red[256];
    float ss = 0.0f;
    for (int j = threadIdx.x; j < 512; j += 256) { float v = x[j]; ss += v * v; }
    red[threadIdx.x] = ss;
    __syncthreads();
    for (int st = 128; st > 0; st >>= 1) {
        if (threadIdx.x < st) red[threadIdx.x] += red[threadIdx.x + st];
        __syncthreads();
    }
    float r = rsqrtf(red[0] * (1.0f / 512.0f) + 1e-6f);
    for (int j = threadIdx.x; j < 512; j += 256)
        out[(size_t)row * 512 + j] = f32_bf16(x[j] * r * w[j]);
}

// ---------------------------------------------------------------------------
// Pack per-head Q (rope on dims 64..127, *1/sqrt(128)), K (k_nope|rope(k_pe)),
// and V^T. Block = 128 thr (=d).
// ---------------------------------------------------------------------------
__launch_bounds__(128)
__global__ void pack_qkv(const float* __restrict__ q, const float* __restrict__ kv,
                         const float* __restrict__ dec,
                         unsigned short* __restrict__ Qh, unsigned short* __restrict__ Kh,
                         unsigned short* __restrict__ VhT, int S, int H) {
    const int d = threadIdx.x;
    const int s = blockIdx.x;
    const int h = blockIdx.y;
    const int b = blockIdx.z;
    const size_t row = (size_t)b * S + s;
    const size_t bh  = (size_t)b * H + h;
    const float scale = 0.08838834764831845f;   // 128^-0.5 folded into Q

    float c = 0.f, sn = 0.f;
    if (d >= 64) {
        int i = (d - 64) >> 1;
        float fr  = __powf(10000.0f, -((float)(2 * i)) / 64.0f);
        float ang = (float)s * fr;
        c  = __cosf(ang);
        sn = __sinf(ang);
    }

    const float* qb = q + row * 2048 + (size_t)h * 128;
    float qv;
    if (d < 64) {
        qv = qb[d];
    } else {
        int i = (d - 64) >> 1;
        float x1 = qb[64 + 2 * i], x2 = qb[64 + 2 * i + 1];
        qv = (d & 1) ? (x2 * c + x1 * sn) : (x1 * c - x2 * sn);
    }
    Qh[(bh * S + s) * 128 + d] = f32_bf16(qv * scale);

    const float* kb = dec + row * 3072 + (size_t)h * 192;
    float kvl;
    if (d < 64) {
        kvl = kb[d];
    } else {
        const float* pe = kv + row * 576 + 512;
        int i = (d - 64) >> 1;
        float x1 = pe[2 * i], x2 = pe[2 * i + 1];
        kvl = (d & 1) ? (x2 * c + x1 * sn) : (x1 * c - x2 * sn);
    }
    Kh[(bh * S + s) * 128 + d] = f32_bf16(kvl);

    VhT[(bh * 128 + d) * S + s] = f32_bf16(kb[64 + d]);
}

// ---------------------------------------------------------------------------
// Flash-style causal attention per (b, h, 64-row q tile). 128 thr = 4 waves;
// wave w owns q rows [16w, 16w+16). Online softmax; WMMA for QK^T and PV.
// ---------------------------------------------------------------------------
#define A_LDQ 65   // dword stride of padded 128-wide bf16 row (130 u16)
#define A_LDV 33   // dword stride of padded 64-wide bf16 row (66 u16)
#define A_LDP 33

__launch_bounds__(128)
__global__ void mla_attention(const unsigned short* __restrict__ Qh,
                              const unsigned short* __restrict__ Kh,
                              const unsigned short* __restrict__ VhT,
                              unsigned short* __restrict__ Out,   // (B*S, H*128) bf16
                              int S, int H) {
    __shared__ unsigned int sQ[64 * A_LDQ];        // [qrow][d]
    __shared__ unsigned int sK[64 * A_LDQ];        // [key][d]
    __shared__ unsigned int sV[128 * A_LDV];       // [d][key]  (V^T)
    __shared__ unsigned short sP[4][16 * (2 * A_LDP)];  // per-wave P 16x64

    const int t  = threadIdx.x;
    const int qt = blockIdx.x;
    const int h  = blockIdx.y;
    const int b  = blockIdx.z;
    const int q0 = qt * 64;
    const size_t bh = (size_t)b * H + h;

    const int wv  = t >> 5;
    const int ln  = t & 31;
    const int lg  = ln >> 4;
    const int l16 = ln & 15;
    const int m0r = 16 * wv;

    // ---- stage Q tile (64x128 bf16) ----
#if USE_TDM
    if (wv == 0) {
        tdm_load_2d(Qh + (bh * S + q0) * 128, lds_byte_off(sQ),
                    /*tile*/ 128u, 64u,
                    /*tensor*/ 128u, (unsigned)(S - q0),
                    128ull, /*pad: 64 dw rows*/ 5u);
    }
#else
#pragma unroll
    for (int it = 0; it < 8; ++it) {
        int ch = it * 128 + t;
        int r  = ch >> 4;
        int c8 = (ch & 15) * 8;
        uint4 d = *(const uint4*)(Qh + (bh * S + q0 + r) * 128 + c8);
        unsigned int* s = &sQ[r * A_LDQ + (c8 >> 1)];
        s[0] = d.x; s[1] = d.y; s[2] = d.z; s[3] = d.w;
    }
#endif

    float mrow[8], lrow[8];
    v8f Oacc[8];
#pragma unroll
    for (int r = 0; r < 8; ++r) { mrow[r] = -1e30f; lrow[r] = 0.0f; }
#pragma unroll
    for (int nt = 0; nt < 8; ++nt)
#pragma unroll
        for (int r = 0; r < 8; ++r) Oacc[nt][r] = 0.0f;

    const int nkt = qt + 1;   // causal
    for (int kt = 0; kt < nkt; ++kt) {
        const int k0 = kt * 64;
        // ---- stage K (64x128) and V^T (128x64) ----
#if USE_TDM
        if (wv == 0) {
            tdm_load_2d(Kh + (bh * S + k0) * 128, lds_byte_off(sK),
                        128u, 64u, 128u, (unsigned)(S - k0), 128ull, 5u);
            tdm_load_2d(VhT + (size_t)bh * 128 * S + k0, lds_byte_off(sV),
                        64u, 128u, (unsigned)(S - k0), 128u,
                        (unsigned long long)S, /*pad: 32 dw rows*/ 4u);
            __builtin_amdgcn_s_wait_tensorcnt(0);   // covers Q on first iter too
        }
#else
#pragma unroll
        for (int it = 0; it < 8; ++it) {
            int ch = it * 128 + t;
            int r  = ch >> 4;
            int c8 = (ch & 15) * 8;
            uint4 d = *(const uint4*)(Kh + (bh * S + k0 + r) * 128 + c8);
            unsigned int* s = &sK[r * A_LDQ + (c8 >> 1)];
            s[0] = d.x; s[1] = d.y; s[2] = d.z; s[3] = d.w;
        }
#pragma unroll
        for (int it = 0; it < 8; ++it) {
            int ch = it * 128 + t;
            int dd = ch >> 3;
            int c8 = (ch & 7) * 8;
            uint4 d = *(const uint4*)(VhT + (bh * 128 + dd) * S + k0 + c8);
            unsigned int* s = &sV[dd * A_LDV + (c8 >> 1)];
            s[0] = d.x; s[1] = d.y; s[2] = d.z; s[3] = d.w;
        }
#endif
        __syncthreads();

        // ---- scores S = Q K^T : 16 rows x 64 keys per wave ----
        v8f sacc[4];
#pragma unroll
        for (int nt = 0; nt < 4; ++nt)
#pragma unroll
            for (int r = 0; r < 8; ++r) sacc[nt][r] = 0.0f;

#pragma unroll
        for (int kk = 0; kk < 4; ++kk) {
            v16bf aq;
            const unsigned int* abase = &sQ[(m0r + l16) * A_LDQ + kk * 16 + lg * 8];
#pragma unroll
            for (int j = 0; j < 8; ++j) {
                int kd = j;
                put2(aq, j, abase[kd]);
            }
#pragma unroll
            for (int nt = 0; nt < 4; ++nt) {
                v16bf bk;
                const unsigned int* bbase =
                    &sK[(nt * 16 + l16) * A_LDQ + kk * 16 + lg * 8];
#pragma unroll
                for (int j = 0; j < 8; ++j) put2(bk, j, bbase[j]);
                sacc[nt] = __builtin_amdgcn_wmma_f32_16x16x32_bf16(
                    false, aq, false, bk, (short)0, sacc[nt], false, false);
            }
        }

        // ---- causal mask ----
#pragma unroll
        for (int nt = 0; nt < 4; ++nt)
#pragma unroll
            for (int r = 0; r < 8; ++r) {
                int qrow = q0 + m0r + lg * 8 + r;
                int kcol = k0 + nt * 16 + l16;
                if (kcol > qrow) sacc[nt][r] = -1e30f;
            }

        // ---- online softmax (row reductions across 16-lane half) ----
#pragma unroll
        for (int r = 0; r < 8; ++r) {
            float rm = -1e30f;
#pragma unroll
            for (int nt = 0; nt < 4; ++nt) rm = fmaxf(rm, sacc[nt][r]);
#pragma unroll
            for (int off = 1; off < 16; off <<= 1)
                rm = fmaxf(rm, __shfl_xor(rm, off, 32));
            float mnew = fmaxf(mrow[r], rm);
            float corr = __expf(mrow[r] - mnew);
            mrow[r] = mnew;
            float ps = 0.0f;
#pragma unroll
            for (int nt = 0; nt < 4; ++nt) {
                float p = __expf(sacc[nt][r] - mnew);
                sacc[nt][r] = p;
                ps += p;
            }
#pragma unroll
            for (int off = 1; off < 16; off <<= 1) ps += __shfl_xor(ps, off, 32);
            lrow[r] = lrow[r] * corr + ps;
#pragma unroll
            for (int nt2 = 0; nt2 < 8; ++nt2) Oacc[nt2][r] *= corr;
        }

        // ---- stage P (C-layout -> A-layout) ----
        unsigned short* pw = &sP[wv][0];
#pragma unroll
        for (int nt = 0; nt < 4; ++nt)
#pragma unroll
            for (int r = 0; r < 8; ++r)
                pw[(lg * 8 + r) * (2 * A_LDP) + nt * 16 + l16] =
                    f32_bf16(sacc[nt][r]);
        __syncthreads();

        // ---- O += P @ V ----
        const unsigned int* pwd = (const unsigned int*)&sP[wv][0];
#pragma unroll
        for (int kk = 0; kk < 2; ++kk) {
            v16bf ap;
            const unsigned int* abase = &pwd[l16 * A_LDP + kk * 16 + lg * 8];
#pragma unroll
            for (int j = 0; j < 8; ++j) {
                int kd = j;
                put2(ap, j, abase[kd]);
            }
#pragma unroll
            for (int nt2 = 0; nt2 < 8; ++nt2) {
                v16bf bv;
                const unsigned int* bbase =
                    &sV[(nt2 * 16 + l16) * A_LDV + kk * 16 + lg * 8];
#pragma unroll
                for (int j = 0; j < 8; ++j) put2(bv, j, bbase[j]);
                Oacc[nt2] = __builtin_amdgcn_wmma_f32_16x16x32_bf16(
                    false, ap, false, bv, (short)0, Oacc[nt2], false, false);
            }
        }
        __syncthreads();
    }

    // ---- normalize and store (bf16, (B*S, H*128)) ----
#pragma unroll
    for (int nt2 = 0; nt2 < 8; ++nt2)
#pragma unroll
        for (int r = 0; r < 8; ++r) {
            int row = q0 + m0r + lg * 8 + r;
            int col = nt2 * 16 + l16;
            float o = Oacc[nt2][r] / lrow[r];
            Out[((size_t)b * S + row) * 2048 + (size_t)h * 128 + col] = f32_bf16(o);
        }
}

// ---------------------------------------------------------------------------
// Host-side orchestration
// ---------------------------------------------------------------------------
extern "C" void kernel_launch(void* const* d_in, const int* in_sizes, int n_in,
                              void* d_out, int out_size, void* d_ws, size_t ws_size,
                              hipStream_t stream) {
    const float* x  = (const float*)d_in[0];
    const float* wq = (const float*)d_in[1];
    const float* wa = (const float*)d_in[2];
    const float* wb = (const float*)d_in[3];
    const float* wn = (const float*)d_in[4];
    const float* wo = (const float*)d_in[5];
    float* out = (float*)d_out;

    const int Bb = 2, S = 2048, H = 16;
    const int M = Bb * S;                 // 4096
    const int DIN = 2048, NQ = 2048, NA = 576, RANK = 512, NDEC = 3072, NOUT = 2048;

    char* ws = (char*)d_ws;
    size_t off = 0;
    auto alloc = [&](size_t bytes) -> void* {
        void* p = ws + off;
        off = (off + bytes + 255) & ~(size_t)255;
        return p;
    };

    unsigned short* x16   = (unsigned short*)alloc((size_t)M * DIN * 2);
    unsigned short* wq16  = (unsigned short*)alloc((size_t)DIN * NQ * 2);
    unsigned short* wa16  = (unsigned short*)alloc((size_t)DIN * NA * 2);
    unsigned short* wb16  = (unsigned short*)alloc((size_t)RANK * NDEC * 2);
    unsigned short* wo16  = (unsigned short*)alloc((size_t)NQ * NOUT * 2);
    float*          qf    = (float*)alloc((size_t)M * NQ * 4);
    float*          kvf   = (float*)alloc((size_t)M * NA * 4);
    unsigned short* kvn16 = (unsigned short*)alloc((size_t)M * RANK * 2);
    float*          dec   = (float*)alloc((size_t)M * NDEC * 4);
    unsigned short* Qh    = (unsigned short*)alloc((size_t)Bb * H * S * 128 * 2);
    unsigned short* Kh    = (unsigned short*)alloc((size_t)Bb * H * S * 128 * 2);
    unsigned short* VhT   = (unsigned short*)alloc((size_t)Bb * H * S * 128 * 2);
    unsigned short* ao16  = (unsigned short*)alloc((size_t)M * NQ * 2);

    auto cvtl = [&](const float* in, unsigned short* o, size_t n) {
        cvt_bf16<<<dim3((unsigned)((n + 255) / 256)), 256, 0, stream>>>(in, o, n);
    };
    cvtl(x,  x16,  (size_t)M * DIN);
    cvtl(wq, wq16, (size_t)DIN * NQ);
    cvtl(wa, wa16, (size_t)DIN * NA);
    cvtl(wb, wb16, (size_t)RANK * NDEC);
    cvtl(wo, wo16, (size_t)NQ * NOUT);

    gemm_bf16_wmma<<<dim3((NQ + 127) / 128, (M + 127) / 128), 256, 0, stream>>>(
        x16, wq16, qf, M, NQ, DIN);
    gemm_bf16_wmma<<<dim3((NA + 127) / 128, (M + 127) / 128), 256, 0, stream>>>(
        x16, wa16, kvf, M, NA, DIN);
    rmsnorm_kv<<<dim3(M), 256, 0, stream>>>(kvf, wn, kvn16);
    gemm_bf16_wmma<<<dim3((NDEC + 127) / 128, (M + 127) / 128), 256, 0, stream>>>(
        kvn16, wb16, dec, M, NDEC, RANK);
    pack_qkv<<<dim3(S, H, Bb), 128, 0, stream>>>(qf, kvf, dec, Qh, Kh, VhT, S, H);
    mla_attention<<<dim3(S / 64, H, Bb), 128, 0, stream>>>(Qh, Kh, VhT, ao16, S, H);
    gemm_bf16_wmma<<<dim3((NOUT + 127) / 128, (M + 127) / 128), 256, 0, stream>>>(
        ao16, wo16, out, M, NOUT, NQ);
}
